// MultiheadAttention_39221641347305
// MI455X (gfx1250) — matrix operand order repaired and checked by reading, not compiled, not run on hardware
//
#include <hip/hip_runtime.h>
#include <hip/hip_bf16.h>

// ---------------------------------------------------------------------------
// Types for CDNA5 WMMA (wave32): v_wmma_f32_16x16x32_bf16
// ---------------------------------------------------------------------------
typedef __attribute__((ext_vector_type(16))) __bf16 bf16x16;
typedef __attribute__((ext_vector_type(8)))  float  f32x8;
typedef __attribute__((ext_vector_type(4)))  int    v4i;

union Frag {
    bf16x16 v;
    uint4   q[2];
};

static __device__ __forceinline__ unsigned short f32_to_bf16(float f) {
    union { float f; unsigned u; } x{f};
    unsigned r = x.u + 0x7FFFu + ((x.u >> 16) & 1u); // round-to-nearest-even
    return (unsigned short)(r >> 16);
}

// ---------------------------------------------------------------------------
// gfx1250 async global->LDS (ASYNCcnt) with compile-safe fallback
// ---------------------------------------------------------------------------
#if defined(__HIP_DEVICE_COMPILE__) && __has_builtin(__builtin_amdgcn_global_load_async_to_lds_b128)
#define HAVE_ASYNC_LDS 1
#else
#define HAVE_ASYNC_LDS 0
#endif

static __device__ __forceinline__ void copy16_g2l(void* lds, const void* g) {
#if HAVE_ASYNC_LDS
    // Builtin expects (v4i addrspace(1)*, v4i addrspace(3)*, imm offset, imm cpol).
    void* gnc = (void*)g;  // drop const
    __attribute__((address_space(1))) char* g1 =
        (__attribute__((address_space(1))) char*)gnc;          // flat -> global
    __attribute__((address_space(3))) char* l3 =
        (__attribute__((address_space(3))) char*)lds;          // flat -> LDS
    __builtin_amdgcn_global_load_async_to_lds_b128(
        (__attribute__((address_space(1))) v4i*)g1,
        (__attribute__((address_space(3))) v4i*)l3, 0, 0);
#else
    *(uint4*)lds = *(const uint4*)g;
#endif
}

static __device__ __forceinline__ void async_wait0() {
#if HAVE_ASYNC_LDS
#if __has_builtin(__builtin_amdgcn_s_wait_asynccnt)
    __builtin_amdgcn_s_wait_asynccnt(0);
#else
    asm volatile("s_wait_asynccnt 0x0" ::: "memory");
#endif
#endif
}

// ---------------------------------------------------------------------------
// fp32 -> bf16 cast (vectorized x4)
// ---------------------------------------------------------------------------
__global__ __launch_bounds__(256) void cast_f32_bf16(const float* __restrict__ in,
                                                     unsigned short* __restrict__ out,
                                                     int n4) {
    int i = blockIdx.x * blockDim.x + threadIdx.x;
    if (i < n4) {
        float4 f = ((const float4*)in)[i];
        ushort4 o;
        o.x = f32_to_bf16(f.x);
        o.y = f32_to_bf16(f.y);
        o.z = f32_to_bf16(f.z);
        o.w = f32_to_bf16(f.w);
        ((ushort4*)out)[i] = o;
    }
}

// ---------------------------------------------------------------------------
// GEMM: C[M,N] = scale * (A[M,K] @ B[N,K]^T), A/B bf16, C bf16 or f32.
// Block tile 128x128, 8 waves (4 along M x 2 along N), wave tile 32x64.
// K stepped by 32; double-buffered LDS fed by async global->LDS copies.
// ---------------------------------------------------------------------------
__global__ __launch_bounds__(256) void gemm_bf16_wmma(
    const unsigned short* __restrict__ A,
    const unsigned short* __restrict__ B,
    void* __restrict__ Cout,
    int M, int N, int K, float scale, int out_bf16)
{
    __shared__ unsigned short sA[2][128 * 40];
    __shared__ unsigned short sB[2][128 * 40];

    const int t    = threadIdx.x;
    const int lane = t & 31;
    const int w    = t >> 5;
    const int wm   = w & 3;    // wave M group: rows wm*32 .. +31
    const int wn   = w >> 2;   // wave N group: cols wn*64 .. +63
    const int lg   = lane >> 4;
    const int ln   = lane & 15;
    const int m0   = blockIdx.y * 128;
    const int n0   = blockIdx.x * 128;

    auto stage = [&](int buf, int kt) {
#pragma unroll
        for (int i = 0; i < 2; ++i) {
            int e = t * 8 + i * 2048;       // 4096 halves per 128x32 tile
            int r = e >> 5;
            int c = e & 31;
            copy16_g2l(&sA[buf][r * 40 + c], &A[(size_t)(m0 + r) * K + kt + c]);
            copy16_g2l(&sB[buf][r * 40 + c], &B[(size_t)(n0 + r) * K + kt + c]);
        }
    };

    f32x8 acc[2][4];
#pragma unroll
    for (int i = 0; i < 2; ++i)
#pragma unroll
        for (int j = 0; j < 4; ++j) acc[i][j] = f32x8{};

    const int nk = K >> 5;
    stage(0, 0);                            // pipeline prologue

    for (int it = 0; it < nk; ++it) {
        const int b = it & 1;
        async_wait0();                      // this wave's copies into buf b done
        __syncthreads();                    // => all waves' copies done
        if (it + 1 < nk) stage(b ^ 1, (it + 1) * 32);  // overlap with compute

        Frag af[2], bf[4];
#pragma unroll
        for (int mi = 0; mi < 2; ++mi) {
            int row = wm * 32 + mi * 16 + ln;
            int bk  = lg * 8;
            af[mi].q[0] = *(const uint4*)&sA[b][row * 40 + bk];       // K {0-7|8-15}
            af[mi].q[1] = *(const uint4*)&sA[b][row * 40 + bk + 16];  // K {16-23|24-31}
        }
#pragma unroll
        for (int ni = 0; ni < 4; ++ni) {
            int row = wn * 64 + ni * 16 + ln;
            int bk  = lg * 8;
            bf[ni].q[0] = *(const uint4*)&sB[b][row * 40 + bk];       // K {0-7|8-15}
            bf[ni].q[1] = *(const uint4*)&sB[b][row * 40 + bk + 16];  // K {16-23|24-31}
        }
#pragma unroll
        for (int mi = 0; mi < 2; ++mi)
#pragma unroll
            for (int ni = 0; ni < 4; ++ni)
                acc[mi][ni] = __builtin_amdgcn_wmma_f32_16x16x32_bf16(
                    false, af[mi].v, false, bf[ni].v, (short)0, acc[mi][ni], false, false);
    }

#pragma unroll
    for (int mi = 0; mi < 2; ++mi)
#pragma unroll
        for (int ni = 0; ni < 4; ++ni)
#pragma unroll
            for (int v = 0; v < 8; ++v) {
                int row = m0 + wm * 32 + mi * 16 + v + lg * 8;
                int col = n0 + wn * 64 + ni * 16 + ln;
                float val = acc[mi][ni][v] * scale;
                if (out_bf16)
                    ((unsigned short*)Cout)[(size_t)row * N + col] = f32_to_bf16(val);
                else
                    ((float*)Cout)[(size_t)row * N + col] = val;
            }
}

// ---------------------------------------------------------------------------
// Flash attention: one head per blockIdx.y, 64 queries per block (4 waves,
// 16 queries/wave), head_dim = 128. Streams 32-key blocks with online softmax.
// K staged via async global->LDS (double-buffered); V prefetched to registers
// one block ahead and stored transposed [d][key]; both overlap the WMMAs.
// Q (pre-scaled by 1/sqrt(128)) / K / V are bf16 [seq, 16*128].
// ---------------------------------------------------------------------------
__global__ __launch_bounds__(128) void flash_attn_wmma(
    const unsigned short* __restrict__ Q,
    const unsigned short* __restrict__ Kd,
    const unsigned short* __restrict__ Vd,
    unsigned short* __restrict__ O,
    int seq)
{
    __shared__ unsigned short sK[2][32 * 136];   // [key][d], stride 136 halves
    __shared__ unsigned short sV[2][128 * 40];   // [d][key], stride 40 halves
    __shared__ unsigned short sP[4][16 * 40];    // wave-private P scratch

    const int t    = threadIdx.x;
    const int lane = t & 31;
    const int w    = t >> 5;
    const int lg   = lane >> 4;
    const int ln   = lane & 15;
    const int h    = blockIdx.y;
    const int hoff = h * 128;
    const int q0   = blockIdx.x * 64 + w * 16;

    uint4 vreg[4];
    auto loadV = [&](int j) {
#pragma unroll
        for (int i = 0; i < 4; ++i) {
            int e = t * 8 + i * 1024;
            int key = e >> 7, d = e & 127;
            vreg[i] = *(const uint4*)&Vd[(size_t)(j + key) * 2048 + hoff + d];
        }
    };
    auto issueK = [&](int buf, int j) {
#pragma unroll
        for (int i = 0; i < 4; ++i) {
            int e = t * 8 + i * 1024;
            int key = e >> 7, d = e & 127;
            copy16_g2l(&sK[buf][key * 136 + d],
                       &Kd[(size_t)(j + key) * 2048 + hoff + d]);
        }
    };

    // Q fragments for this wave's 16 query rows: 4 d-chunks of 32
    Frag qf[4];
    {
        const unsigned short* qrow = Q + (size_t)(q0 + ln) * 2048 + hoff;
        int bk = lg * 8;
#pragma unroll
        for (int c = 0; c < 4; ++c) {
            qf[c].q[0] = *(const uint4*)&qrow[c * 32 + bk];
            qf[c].q[1] = *(const uint4*)&qrow[c * 32 + bk + 16];
        }
    }

    f32x8 o[8];
#pragma unroll
    for (int i = 0; i < 8; ++i) o[i] = f32x8{};
    float mrow[8], lrow[8];
#pragma unroll
    for (int v = 0; v < 8; ++v) { mrow[v] = -1e30f; lrow[v] = 0.0f; }

    issueK(0, 0);      // pipeline prologue
    loadV(0);

    for (int j = 0; j < seq; j += 32) {
        const int b = (j >> 5) & 1;
        async_wait0();                       // K(j) resident in sK[b]
        // store current V block transposed into sV[b]
#pragma unroll
        for (int i = 0; i < 4; ++i) {
            int e = t * 8 + i * 1024;
            int key = e >> 7, d = e & 127;
            union { uint4 q; unsigned short s[8]; } vv;
            vv.q = vreg[i];
#pragma unroll
            for (int x = 0; x < 8; ++x) sV[b][(d + x) * 40 + key] = vv.s[x];
        }
        __syncthreads();
        if (j + 32 < seq) {                  // overlap next block's fetch
            issueK(b ^ 1, j + 32);
            loadV(j + 32);
        }

        // S = Q K^T : 16 queries x 32 keys (2 N-subtiles, 4 d-chunks each)
        f32x8 s[2];
        s[0] = f32x8{}; s[1] = f32x8{};
#pragma unroll
        for (int tn = 0; tn < 2; ++tn) {
#pragma unroll
            for (int c = 0; c < 4; ++c) {
                Frag kf;
                int base = (tn * 16 + ln) * 136 + c * 32 + lg * 8;
                kf.q[0] = *(const uint4*)&sK[b][base];
                kf.q[1] = *(const uint4*)&sK[b][base + 16];
                s[tn] = __builtin_amdgcn_wmma_f32_16x16x32_bf16(
                    false, qf[c].v, false, kf.v, (short)0, s[tn], false, false);
            }
        }

        // online softmax (rows m = v + 8*lg live across the 16 lanes of a half)
        float alpha[8];
#pragma unroll
        for (int v = 0; v < 8; ++v) {
            float rm = fmaxf(s[0][v], s[1][v]);
#pragma unroll
            for (int off = 8; off >= 1; off >>= 1)
                rm = fmaxf(rm, __shfl_xor(rm, off, 32));
            float mn = fmaxf(mrow[v], rm);
            alpha[v] = __expf(mrow[v] - mn);
            float p0 = __expf(s[0][v] - mn);
            float p1 = __expf(s[1][v] - mn);
            float rs = p0 + p1;
#pragma unroll
            for (int off = 8; off >= 1; off >>= 1)
                rs += __shfl_xor(rs, off, 32);
            lrow[v] = lrow[v] * alpha[v] + rs;
            mrow[v] = mn;
            s[0][v] = p0;
            s[1][v] = p1;
        }
#pragma unroll
        for (int tn = 0; tn < 8; ++tn)
#pragma unroll
            for (int v = 0; v < 8; ++v) o[tn][v] *= alpha[v];

        // P: C-layout -> bf16 A-layout via wave-private LDS (DS in-order per wave)
        unsigned short* P = sP[w];
#pragma unroll
        for (int tn = 0; tn < 2; ++tn)
#pragma unroll
            for (int v = 0; v < 8; ++v)
                P[(v + lg * 8) * 40 + tn * 16 + ln] = f32_to_bf16(s[tn][v]);
        asm volatile("" ::: "memory");

        Frag pf;
        {
            int bk = lg * 8;
            pf.q[0] = *(const uint4*)&P[ln * 40 + bk];
            pf.q[1] = *(const uint4*)&P[ln * 40 + bk + 16];
        }

        // O += P V : 8 N-subtiles over head_dim
#pragma unroll
        for (int tn = 0; tn < 8; ++tn) {
            Frag vf;
            int base = (tn * 16 + ln) * 40 + lg * 8;
            vf.q[0] = *(const uint4*)&sV[b][base];
            vf.q[1] = *(const uint4*)&sV[b][base + 16];
            o[tn] = __builtin_amdgcn_wmma_f32_16x16x32_bf16(
                false, pf.v, false, vf.v, (short)0, o[tn], false, false);
        }
    }

    // normalize and emit bf16 attn output [seq, 16*128]
#pragma unroll
    for (int tn = 0; tn < 8; ++tn)
#pragma unroll
        for (int v = 0; v < 8; ++v) {
            int row = q0 + v + lg * 8;
            int col = hoff + tn * 16 + ln;
            O[(size_t)row * 2048 + col] = f32_to_bf16(o[tn][v] / lrow[v]);
        }
}

// ---------------------------------------------------------------------------
// Host orchestration
// ---------------------------------------------------------------------------
extern "C" void kernel_launch(void* const* d_in, const int* in_sizes, int n_in,
                              void* d_out, int out_size, void* d_ws, size_t ws_size,
                              hipStream_t stream) {
    const int SEQ = 2048, DM = 2048;
    const size_t SLAB = (size_t)SEQ * DM;           // elements per matrix
    const size_t SLABB = SLAB * sizeof(unsigned short);

    const float* x  = (const float*)d_in[0];
    const float* Wq = (const float*)d_in[1];
    const float* Wk = (const float*)d_in[2];
    const float* Wv = (const float*)d_in[3];
    const float* Wo = (const float*)d_in[4];

    char* ws = (char*)d_ws;
    unsigned short* Xb  = (unsigned short*)(ws + 0 * SLABB);
    unsigned short* Wqb = (unsigned short*)(ws + 1 * SLABB);
    unsigned short* Wkb = (unsigned short*)(ws + 2 * SLABB);
    unsigned short* Wvb = (unsigned short*)(ws + 3 * SLABB);
    unsigned short* Wob = (unsigned short*)(ws + 4 * SLABB);
    unsigned short* Qb  = (unsigned short*)(ws + 5 * SLABB);
    unsigned short* Kb  = (unsigned short*)(ws + 6 * SLABB);
    unsigned short* Vb  = (unsigned short*)(ws + 7 * SLABB);
    unsigned short* Ab  = (unsigned short*)(ws + 8 * SLABB);

    const int n4 = (int)(SLAB / 4);
    const int cg = (n4 + 255) / 256;
    cast_f32_bf16<<<cg, 256, 0, stream>>>(x,  Xb,  n4);
    cast_f32_bf16<<<cg, 256, 0, stream>>>(Wq, Wqb, n4);
    cast_f32_bf16<<<cg, 256, 0, stream>>>(Wk, Wkb, n4);
    cast_f32_bf16<<<cg, 256, 0, stream>>>(Wv, Wvb, n4);
    cast_f32_bf16<<<cg, 256, 0, stream>>>(Wo, Wob, n4);

    dim3 gg(DM / 128, SEQ / 128);
    const float qscale = 0.08838834764831845f;  // 1/sqrt(128)
    gemm_bf16_wmma<<<gg, 256, 0, stream>>>(Xb, Wqb, Qb, SEQ, DM, DM, qscale, 1);
    gemm_bf16_wmma<<<gg, 256, 0, stream>>>(Xb, Wkb, Kb, SEQ, DM, DM, 1.0f, 1);
    gemm_bf16_wmma<<<gg, 256, 0, stream>>>(Xb, Wvb, Vb, SEQ, DM, DM, 1.0f, 1);

    flash_attn_wmma<<<dim3(SEQ / 64, 16), 128, 0, stream>>>(Qb, Kb, Vb, Ab, SEQ);

    gemm_bf16_wmma<<<gg, 256, 0, stream>>>(Ab, Wob, (float*)d_out, SEQ, DM, DM, 1.0f, 0);
}
